// Decoder_75230647157279
// MI455X (gfx1250) — matrix operand order repaired and checked by reading, not compiled, not run on hardware
//
#include <hip/hip_runtime.h>
#include <hip/hip_bf16.h>
#include <math.h>

#define BB   64
#define TT   1024
#define HH   512
#define EE   128
#define SS   128
#define II   32
#define NG   2048      // 4*H
#define KX   1664      // E + 3*H (W_ih row length)
#define KC   1024      // merged K: 512 (h) + 512 (enc)
#define NWG  32        // persistent workgroups
#define NTH  256       // 8 waves
#define HS   16        // h columns owned per WG

typedef __attribute__((ext_vector_type(16))) __bf16 v16bf;
typedef __attribute__((ext_vector_type(8)))  float  v8f;

union ABReg { uint4 q[2]; v16bf v; __bf16 e[16]; };

__device__ __forceinline__ float sigf(float x){ return 1.0f/(1.0f+__expf(-x)); }

#define WMMA_BF16(A,Bv,C) __builtin_amdgcn_wmma_f32_16x16x32_bf16(false,(A),false,(Bv),(short)0,(C),false,false)

// B operand from LDS: tile = 32 lanes x 16 bf16 contiguous (1KB)
__device__ __forceinline__ v16bf fetch_b_lds(const __bf16* lds, int lt, int lane){
  const uint4* p = (const uint4*)(lds + (size_t)lt * 512) + lane * 2;
  ABReg r; r.q[0] = p[0]; r.q[1] = p[1];
  return r.v;
}

// A operand from row-major bf16, arbitrary row stride:
// lane<16 holds row m0+lane, K {k0+0..15}; lanes 16-31: K {k0+16..31}
__device__ __forceinline__ v16bf fetch_a_bfs(const __bf16* __restrict__ base, size_t rs,
                                             int m0, int k0, int lane){
  int lr = lane & 15, hi = lane >> 4;
  const __bf16* p = base + (size_t)(m0 + lr) * rs + k0 + hi * 16;
  ABReg r;
  r.q[0] = *(const uint4*)(p);
  r.q[1] = *(const uint4*)(p + 8);
  return r.v;
}

__device__ __forceinline__ void grid_barrier(unsigned* syncw, unsigned gen){
  __threadfence();
  __syncthreads();
  if (threadIdx.x == 0){
    if (atomicAdd(&syncw[0], 1u) == (unsigned)(NWG - 1)){
      atomicExch(&syncw[0], 0u);
      __threadfence();
      atomicAdd(&syncw[1], 1u);
    } else {
      while (atomicAdd(&syncw[1], 0u) == gen) { __builtin_amdgcn_s_sleep(1); }
    }
  }
  __syncthreads();
}

// ---------------- attention logits: one wave per (b,t) ----------------
__global__ void att_logits_kernel(const float* __restrict__ enc,
                                  const float* __restrict__ att_w,
                                  const float* __restrict__ att_b,
                                  float* __restrict__ logits){
  int gw   = blockIdx.x * 8 + (threadIdx.x >> 5);   // == b*TT + t
  int lane = threadIdx.x & 31;
  const float* row = enc + (size_t)gw * HH;
  const float* w   = att_w + HH;                     // att_w[H:,0]
  float s = 0.f;
  for (int k = lane * 4; k < HH; k += 128){
    float4 e  = *(const float4*)(row + k);
    float4 ww = *(const float4*)(w + k);
    s += e.x*ww.x + e.y*ww.y + e.z*ww.z + e.w*ww.w;
  }
  for (int off = 16; off; off >>= 1) s += __shfl_xor(s, off, 32);
  if (lane == 0) logits[gw] = s + att_b[0];
}

// ---------------- softmax + context + pooled mean: one block per b ----------------
__global__ void ctx_kernel(const float* __restrict__ enc,
                           const float* __restrict__ logits,
                           const int*   __restrict__ seq_lens,
                           float* __restrict__ context,
                           float* __restrict__ pooled){
  __shared__ float red[NTH];
  int b = blockIdx.x, tid = threadIdx.x;
  int len = seq_lens[b];
  const float* lrow = logits + (size_t)b * TT;
  float mx = -3.0e38f;
  for (int t = tid; t < len; t += NTH) mx = fmaxf(mx, lrow[t]);
  red[tid] = mx; __syncthreads();
  for (int s = NTH/2; s; s >>= 1){ if (tid < s) red[tid] = fmaxf(red[tid], red[tid+s]); __syncthreads(); }
  mx = red[0]; __syncthreads();
  float z = 0.f;
  for (int t = tid; t < len; t += NTH) z += __expf(lrow[t] - mx);
  red[tid] = z; __syncthreads();
  for (int s = NTH/2; s; s >>= 1){ if (tid < s) red[tid] += red[tid+s]; __syncthreads(); }
  z = red[0]; __syncthreads();
  float inv = 1.f / z;
  float c0=0.f, c1=0.f, p0=0.f, p1=0.f;
  for (int t = 0; t < len; ++t){
    float p = __expf(lrow[t] - mx) * inv;
    const float* er = enc + ((size_t)b * TT + t) * HH;
    float e0 = er[tid], e1 = er[tid + NTH];
    c0 += p * e0; c1 += p * e1; p0 += e0; p1 += e1;
  }
  float invl = 1.f / (float)len;
  context[(size_t)b*HH + tid]        = c0;
  context[(size_t)b*HH + tid + NTH]  = c1;
  pooled [(size_t)b*HH + tid]        = p0 * invl;
  pooled [(size_t)b*HH + tid + NTH]  = p1 * invl;
}

// ---------------- enc fp32 -> bf16 (one-time; halves the serial-loop A traffic) ----
__global__ void encbf_kernel(const float* __restrict__ enc, __bf16* __restrict__ enc_bf){
  size_t i = ((size_t)blockIdx.x * NTH + threadIdx.x) * 8;
  if (i >= (size_t)BB * TT * HH) return;
  float4 f0 = *(const float4*)(enc + i);
  float4 f1 = *(const float4*)(enc + i + 4);
  ABReg r;
  r.e[0]=(__bf16)f0.x; r.e[1]=(__bf16)f0.y; r.e[2]=(__bf16)f0.z; r.e[3]=(__bf16)f0.w;
  r.e[4]=(__bf16)f1.x; r.e[5]=(__bf16)f1.y; r.e[6]=(__bf16)f1.z; r.e[7]=(__bf16)f1.w;
  *(uint4*)(enc_bf + i) = r.q[0];
}

// ---------------- weight fold + WMMA-B swizzle ----------------
// mode 0 (Wcc, ntot=2048, ktot=1024): k<512 -> W_ih[:,0:512]+W_hh ; k>=512 -> W_ih[:,640:1152]
// mode 1 (Wslot, ntot=128, ktot=512): W_slot
__global__ void swz_kernel(const float* __restrict__ src0,
                           const float* __restrict__ src1,
                           __bf16* __restrict__ dst, int ntot, int ktot, int mode){
  int tid = blockIdx.x * blockDim.x + threadIdx.x;
  if (tid >= ntot * ktot) return;
  int tile = tid >> 9, r = tid & 511;
  int lane = r >> 4, e = r & 15;
  int ntiles = ntot >> 4;
  int n = (tile % ntiles) * 16 + (lane & 15);
  int k = (tile / ntiles) * 32 + (lane >> 4) * 16 + e;
  float v;
  if (mode == 0){
    if (k < 512) v = src0[(size_t)n*KX + k] + src1[(size_t)n*HH + k];
    else         v = src0[(size_t)n*KX + 640 + (k - 512)];
  } else {
    v = src0[(size_t)n*HH + k];
  }
  dst[tid] = (__bf16)v;
}

// emb_gate[s][n] = emb_row(s) . W_ih[n, 512:640]   (s==SS -> init_slot)
__global__ void embgate_kernel(const float* __restrict__ W_ih,
                               const float* __restrict__ table,
                               const float* __restrict__ init_slot,
                               float* __restrict__ emb_gate){
  int tid = blockIdx.x * blockDim.x + threadIdx.x;
  if (tid >= (SS + 1) * NG) return;
  int s = tid / NG, n = tid % NG;
  const float* row = (s < SS) ? (table + (size_t)s * EE) : init_slot;
  const float* w   = W_ih + (size_t)n * KX + 512;
  float acc = 0.f;
  for (int e2 = 0; e2 < EE; ++e2) acc += row[e2] * w[e2];
  emb_gate[tid] = acc;
}

// bias_all[b][n] = b_ih[n] + b_hh[n] + context[b] . W_ih[n, 1152:1664]
__global__ void biasall_kernel(const float* __restrict__ W_ih,
                               const float* __restrict__ b_ih,
                               const float* __restrict__ b_hh,
                               const float* __restrict__ context,
                               float* __restrict__ bias_all){
  int tid = blockIdx.x * blockDim.x + threadIdx.x;
  if (tid >= BB * NG) return;
  int b = tid / NG, n = tid % NG;
  const float* ctx = context + (size_t)b * HH;
  const float* w   = W_ih + (size_t)n * KX + 1152;
  float acc = b_ih[n] + b_hh[n];
  for (int k = 0; k < HH; ++k) acc += ctx[k] * w[k];
  bias_all[tid] = acc;
}

__global__ void init_kernel(const float* __restrict__ lstm_h,
                            __bf16* __restrict__ hbuf,
                            unsigned* __restrict__ syncw){
  int tid = blockIdx.x * blockDim.x + threadIdx.x;
  if (tid < BB * HH) hbuf[tid] = (__bf16)lstm_h[tid];
  if (tid < 8) syncw[tid] = 0u;
}

// ---------------- persistent LSTM scan: 32 WGs, weights LDS-resident ----------------
__global__ __launch_bounds__(NTH) void scan_kernel(
    const __bf16* __restrict__ enc_bf, const int* __restrict__ seq_lens,
    const float* __restrict__ lstm_h, const float* __restrict__ lstm_c,
    const float* __restrict__ bias_all, const float* __restrict__ emb_gate,
    const float* __restrict__ b_slot,
    const __bf16* __restrict__ Wcc, const __bf16* __restrict__ Wslot,
    __bf16* __restrict__ hbuf, unsigned* __restrict__ syncw,
    float* __restrict__ out_slot)
{
  // LDS: 128KB gate-weight slice + 128KB slot weights + 32KB logits/reduce
  __shared__ __align__(32) __bf16 WccL [128 * 512];   // tiles lt = kt*4+q  (kt 0..31)
  __shared__ __align__(32) __bf16 WslL [128 * 512];   // tiles lt = kt*8+nt (kt 0..15)
  __shared__ __align__(16) float  slotL[BB * SS];     // also Phase-A K-reduction buffer
  __shared__ float lseL[BB];
  __shared__ int   idxL[BB];
  __shared__ int   lenL[BB];

  const int g    = blockIdx.x;
  const int tid  = threadIdx.x;
  const int wave = tid >> 5, lane = tid & 31;
  const int lr   = lane & 15, hi = lane >> 4;
  const int mtile = wave & 3, khalf = wave >> 2;   // Phase A: 4 m-tiles x 2 K-halves
  const int cg   = khalf;                          // Phase B: 2 slot-tile groups
  const int m0 = mtile * 16;
  const int hcol = g * HS + lr;                    // h column owned (khalf==0 lanes)

  // ---- stage weights into LDS (once) ----
  for (int c = tid; c < 128 * 64; c += NTH){       // Wcc slice: 128 tiles x 64 16B-chunks
    int lt = c >> 6, j = c & 63;
    int kt = lt >> 2, q = lt & 3;
    size_t gt = (size_t)kt * (NG >> 4) + (q * 32 + g);
    *(uint4*)(WccL + (size_t)lt * 512 + j * 8) = *(const uint4*)(Wcc + gt * 512 + j * 8);
  }
  for (int c = tid; c < 128 * 64; c += NTH){       // Wslot: linear copy (layout identical)
    *(uint4*)(WslL + (size_t)c * 8) = *(const uint4*)(Wslot + (size_t)c * 8);
  }
  if (tid < BB){ lenL[tid] = seq_lens[tid]; idxL[tid] = SS; }  // SS == init_slot row
  __syncthreads();

  float c_reg[8], h_reg[8];
  #pragma unroll
  for (int r = 0; r < 8; ++r){
    int mb = m0 + hi * 8 + r;
    c_reg[r] = lstm_c[(size_t)mb * HH + hcol];
    h_reg[r] = lstm_h[(size_t)mb * HH + hcol];
  }

  unsigned gen = 0;
  int cur = 0;
  for (int t = 0; t < TT; ++t){
    const __bf16* hsrc = hbuf + (size_t)cur * BB * HH;
    __bf16*       hdst = hbuf + (size_t)(cur ^ 1) * BB * HH;

    // Uniform A source for both K-halves: bf16, only base/stride differ (loop-invariant)
    const __bf16* abase = (khalf == 0) ? hsrc : (enc_bf + (size_t)t * HH);
    const size_t  astr  = (khalf == 0) ? (size_t)HH : (size_t)TT * HH;
    if (khalf == 1 && t < TT - 1)  // prefetch next timestep's enc rows
      __builtin_prefetch(enc_bf + (size_t)(t + 1) * HH + (size_t)(m0 + lr) * TT * HH, 0, 1);

    // ---- Phase A: gates = [h | enc_t] @ Wcc^T ; waves split K (0-3: h, 4-7: enc)
    v8f acc[4] = {};
    {
      const int ktb = khalf * 16;
      v16bf a  = fetch_a_bfs(abase, astr, m0, 0, lane);
      v16bf b0 = fetch_b_lds(WccL, (ktb + 0) * 4 + 0, lane);
      v16bf b1 = fetch_b_lds(WccL, (ktb + 0) * 4 + 1, lane);
      v16bf b2 = fetch_b_lds(WccL, (ktb + 0) * 4 + 2, lane);
      v16bf b3 = fetch_b_lds(WccL, (ktb + 0) * 4 + 3, lane);
      #pragma unroll
      for (int i = 0; i < 16; ++i){
        v16bf an, n0v, n1v, n2v, n3v;
        if (i < 15){
          an  = fetch_a_bfs(abase, astr, m0, (i + 1) * 32, lane);
          n0v = fetch_b_lds(WccL, (ktb + i + 1) * 4 + 0, lane);
          n1v = fetch_b_lds(WccL, (ktb + i + 1) * 4 + 1, lane);
          n2v = fetch_b_lds(WccL, (ktb + i + 1) * 4 + 2, lane);
          n3v = fetch_b_lds(WccL, (ktb + i + 1) * 4 + 3, lane);
        }
        acc[0] = WMMA_BF16(a, b0, acc[0]);
        acc[1] = WMMA_BF16(a, b1, acc[1]);
        acc[2] = WMMA_BF16(a, b2, acc[2]);
        acc[3] = WMMA_BF16(a, b3, acc[3]);
        a = an; b0 = n0v; b1 = n1v; b2 = n2v; b3 = n3v;
      }
    }
    // cross-wave K reduction through LDS (reuse slotL)
    if (khalf == 1){
      #pragma unroll
      for (int q = 0; q < 4; ++q)
        #pragma unroll
        for (int r = 0; r < 8; ++r)
          slotL[(mtile * 4 + q) * 256 + (hi * 8 + r) * 16 + lr] = acc[q][r];
    }
    __syncthreads();
    if (khalf == 0){
      #pragma unroll
      for (int q = 0; q < 4; ++q)
        #pragma unroll
        for (int r = 0; r < 8; ++r)
          acc[q][r] += slotL[(mtile * 4 + q) * 256 + (hi * 8 + r) * 16 + lr];

      #pragma unroll
      for (int r = 0; r < 8; ++r){
        int mb = m0 + hi * 8 + r;
        const float* brow = bias_all + (size_t)mb * NG;
        const float* erow = emb_gate + (size_t)idxL[mb] * NG;
        int nb = g * HS + lr;
        float gi = acc[0][r] + brow[nb]         + erow[nb];
        float gf = acc[1][r] + brow[512 + nb]   + erow[512 + nb];
        float gg = acc[2][r] + brow[1024 + nb]  + erow[1024 + nb];
        float go = acc[3][r] + brow[1536 + nb]  + erow[1536 + nb];
        float cn = sigf(gf) * c_reg[r] + sigf(gi) * tanhf(gg);
        float hn = sigf(go) * tanhf(cn);
        bool v = t < lenL[mb];
        c_reg[r] = v ? cn : c_reg[r];
        h_reg[r] = v ? hn : h_reg[r];
        hdst[(size_t)mb * HH + hcol] = (__bf16)h_reg[r];   // masked carry published
      }
    }

    grid_barrier(syncw, gen); gen++;                        // h fully published

    // ---- Phase B: slot logits (redundant per WG -> local argmax, B from LDS)
    {
      v8f sacc[4] = {};
      v16bf a  = fetch_a_bfs(hdst, HH, m0, 0, lane);
      v16bf b0 = fetch_b_lds(WslL, 0 * 8 + cg * 4 + 0, lane);
      v16bf b1 = fetch_b_lds(WslL, 0 * 8 + cg * 4 + 1, lane);
      v16bf b2 = fetch_b_lds(WslL, 0 * 8 + cg * 4 + 2, lane);
      v16bf b3 = fetch_b_lds(WslL, 0 * 8 + cg * 4 + 3, lane);
      #pragma unroll
      for (int kt = 0; kt < 16; ++kt){
        v16bf an, n0v, n1v, n2v, n3v;
        if (kt < 15){
          an  = fetch_a_bfs(hdst, HH, m0, (kt + 1) * 32, lane);
          n0v = fetch_b_lds(WslL, (kt + 1) * 8 + cg * 4 + 0, lane);
          n1v = fetch_b_lds(WslL, (kt + 1) * 8 + cg * 4 + 1, lane);
          n2v = fetch_b_lds(WslL, (kt + 1) * 8 + cg * 4 + 2, lane);
          n3v = fetch_b_lds(WslL, (kt + 1) * 8 + cg * 4 + 3, lane);
        }
        sacc[0] = WMMA_BF16(a, b0, sacc[0]);
        sacc[1] = WMMA_BF16(a, b1, sacc[1]);
        sacc[2] = WMMA_BF16(a, b2, sacc[2]);
        sacc[3] = WMMA_BF16(a, b3, sacc[3]);
        a = an; b0 = n0v; b1 = n1v; b2 = n2v; b3 = n3v;
      }
      #pragma unroll
      for (int j = 0; j < 4; ++j){
        int n = (cg * 4 + j) * 16 + lr;
        #pragma unroll
        for (int r = 0; r < 8; ++r){
          int mb = m0 + hi * 8 + r;
          slotL[mb * SS + n] = sacc[j][r] + b_slot[n];
        }
      }
    }
    __syncthreads();
    if (tid < BB){
      int b = tid;
      float mx = slotL[b * SS]; int am = 0;
      for (int n = 1; n < SS; ++n){ float x = slotL[b * SS + n]; if (x > mx){ mx = x; am = n; } }
      float z = 0.f;
      for (int n = 0; n < SS; ++n) z += __expf(slotL[b * SS + n] - mx);
      lseL[b] = mx + __logf(z);
      if (t < lenL[b]) idxL[b] = am;                        // emb index for next step
    }
    __syncthreads();
    {
      int b0w = g * (BB / NWG);                             // each WG writes 2 batch rows
      for (int i = tid; i < (BB / NWG) * SS; i += NTH){
        int b = b0w + (i >> 7), n = i & 127;
        bool v = t < lenL[b];
        out_slot[((size_t)b * TT + t) * SS + n] = v ? (slotL[b * SS + n] - lseL[b]) : 0.0f;
      }
    }
    __syncthreads();   // protect slotL before next step's K-reduction reuse
    cur ^= 1;
  }
}

// ---------------- intent head ----------------
__global__ void intent_kernel(const float* __restrict__ context,
                              const float* __restrict__ pooled,
                              const float* __restrict__ W_int,
                              const float* __restrict__ b_int,
                              float* __restrict__ out){
  __shared__ float lg[II];
  __shared__ float lse;
  int b = blockIdx.x, i = threadIdx.x;
  if (i < II){
    const float* w = W_int + (size_t)i * (2 * HH);
    float a = b_int[i];
    for (int k = 0; k < HH; ++k) a += context[(size_t)b*HH + k] * w[k];
    for (int k = 0; k < HH; ++k) a += pooled [(size_t)b*HH + k] * w[HH + k];
    lg[i] = a;
  }
  __syncthreads();
  if (i == 0){
    float mx = lg[0];
    for (int j = 1; j < II; ++j) mx = fmaxf(mx, lg[j]);
    float z = 0.f;
    for (int j = 0; j < II; ++j) z += __expf(lg[j] - mx);
    lse = mx + __logf(z);
  }
  __syncthreads();
  if (i < II) out[(size_t)b * II + i] = lg[i] - lse;
}

extern "C" void kernel_launch(void* const* d_in, const int* in_sizes, int n_in,
                              void* d_out, int out_size, void* d_ws, size_t ws_size,
                              hipStream_t stream){
  const float* lstm_h   = (const float*)d_in[0];
  const float* lstm_c   = (const float*)d_in[1];
  const float* enc      = (const float*)d_in[2];
  const int*   seq_lens = (const int*)  d_in[3];
  const float* att_w    = (const float*)d_in[4];
  const float* att_b    = (const float*)d_in[5];
  const float* W_ih     = (const float*)d_in[6];
  const float* b_ih     = (const float*)d_in[7];
  const float* W_hh     = (const float*)d_in[8];
  const float* b_hh     = (const float*)d_in[9];
  const float* W_slot   = (const float*)d_in[10];
  const float* b_slot   = (const float*)d_in[11];
  const float* emb_tab  = (const float*)d_in[12];
  const float* init_sl  = (const float*)d_in[13];
  const float* W_int    = (const float*)d_in[14];
  const float* b_int    = (const float*)d_in[15];
  float* out = (float*)d_out;

  char* w = (char*)d_ws;
  size_t off = 0;
  auto take = [&](size_t bytes)->void*{
    void* p = w + off; off = (off + bytes + 255) & ~(size_t)255; return p;
  };
  float*    logits   = (float*)   take((size_t)BB*TT*4);
  float*    context  = (float*)   take((size_t)BB*HH*4);
  float*    pooled   = (float*)   take((size_t)BB*HH*4);
  float*    bias_all = (float*)   take((size_t)BB*NG*4);
  float*    emb_gate = (float*)   take((size_t)(SS+1)*NG*4);
  __bf16*   Wcc      = (__bf16*)  take((size_t)NG*KC*2);
  __bf16*   Wslz     = (__bf16*)  take((size_t)SS*HH*2);
  __bf16*   hbuf     = (__bf16*)  take((size_t)2*BB*HH*2);
  unsigned* syncw    = (unsigned*)take(256);
  __bf16*   enc_bf   = (__bf16*)  take((size_t)BB*TT*HH*2);   // 64MB bf16 copy of enc

  att_logits_kernel<<<BB*TT/8, NTH, 0, stream>>>(enc, att_w, att_b, logits);
  ctx_kernel<<<BB, NTH, 0, stream>>>(enc, logits, seq_lens, context, pooled);
  encbf_kernel<<<(int)(((size_t)BB*TT*HH/8 + NTH-1)/NTH), NTH, 0, stream>>>(enc, enc_bf);
  swz_kernel<<<(NG*KC + NTH-1)/NTH, NTH, 0, stream>>>(W_ih, W_hh, Wcc, NG, KC, 0);
  swz_kernel<<<(SS*HH + NTH-1)/NTH, NTH, 0, stream>>>(W_slot, W_slot, Wslz, SS, HH, 1);
  embgate_kernel<<<((SS+1)*NG + NTH-1)/NTH, NTH, 0, stream>>>(W_ih, emb_tab, init_sl, emb_gate);
  biasall_kernel<<<(BB*NG + NTH-1)/NTH, NTH, 0, stream>>>(W_ih, b_ih, b_hh, context, bias_all);
  init_kernel<<<(BB*HH + NTH-1)/NTH, NTH, 0, stream>>>(lstm_h, hbuf, syncw);
  scan_kernel<<<NWG, NTH, 0, stream>>>(enc_bf, seq_lens, lstm_h, lstm_c, bias_all, emb_gate,
                                       b_slot, Wcc, Wslz, hbuf, syncw, out);
  intent_kernel<<<BB, 32, 0, stream>>>(context, pooled, W_int, b_int,
                                       out + (size_t)BB*TT*SS);
}
